// LocalAttention_46986942218547
// MI455X (gfx1250) — compile-verified, matrix operand order checked by reading, not run on hardware
//
#include <hip/hip_runtime.h>
#include <hip/hip_bf16.h>
#include <math.h>

// ---------------------------------------------------------------------------
// Local windowed attention with RoPE for MI455X (gfx1250, wave32, WMMA).
//   B=8 H=4 n=4096 d=64 WINDOW=128 look_backward=1 -> each 128-row query
//   window attends to 256 keys (prev window + own window), causal-masked.
// One workgroup (256 thr = 8 waves) per (batch*head, window) tile.
// bf16 WMMA 16x16x32 with f32 accumulation for both QK^T and P*V.
//  - qk scale folded into Q at RoPE time
//  - native f32->bf16 converts ((__bf16) casts)
//  - V stored transposed in LDS -> all fragments gather with dword loads
//  - P stored unnormalized; 1/rowsum applied at the final output store
// ---------------------------------------------------------------------------

typedef __attribute__((ext_vector_type(16))) __bf16 v16bf;
typedef __attribute__((ext_vector_type(8)))  float  v8f;

#define WIN   128
#define KW    256
#define DIM   64
#define NSEQ  4096
#define NHEAD 4
#define STQ   68      // LDS row stride (bf16 elems) for Q/K, padded
#define STV   258     // LDS row stride for V^T: 516B = 129 dwords -> +1 bank/row
#define STP   260     // LDS row stride for probabilities (256 + pad)

// Fragment gather per ISA 7.12.2 (16-bit, 16x32 A / 32x16 B):
//   lanes 0-15 : K-base 0  (elements e<8 -> K=e,   e>=8 -> K=16+(e-8))
//   lanes 16-31: K-base 8  (elements e<8 -> K=8+e, e>=8 -> K=24+(e-8))
// K-pairs are contiguous in LDS -> dword loads.
__device__ __forceinline__ v16bf gather_row(const __bf16* base, int row,
                                            int stride, int kbase, int laneHalf) {
  union { unsigned int u[8]; v16bf v; } f;
  const __bf16* p = base + row * stride + kbase + laneHalf * 8;
#pragma unroll
  for (int g = 0; g < 2; ++g)
#pragma unroll
    for (int q = 0; q < 4; ++q)
      f.u[g * 4 + q] = *(const unsigned int*)(p + g * 16 + q * 2);
  return f.v;
}

__global__ __launch_bounds__(256)
void LocalAttention_46986942218547_kernel(const float* __restrict__ qg,
                                          const float* __restrict__ kg,
                                          const float* __restrict__ vg,
                                          const int*   __restrict__ amask,
                                          float*       __restrict__ outg) {
  __shared__ __bf16 sQ[WIN * STQ];    // RoPE'd Q * scale, bf16
  __shared__ __bf16 sK[KW  * STQ];    // RoPE'd K (prev+cur window)
  __shared__ __bf16 sVt[DIM * STV];   // V transposed: [dim][key]
  __shared__ __bf16 sP[WIN * STP];    // unnormalized softmax numerators
  __shared__ unsigned short sInval[KW];   // 1 = key invalid (pad / mask==0)

  const int w    = blockIdx.x;        // window index 0..31
  const int bh   = blockIdx.y;        // batch*head 0..31
  const int bidx = bh / NHEAD;        // batch index for attention mask
  const int tid  = threadIdx.x;
  const int lane = tid & 31;
  const int wv   = tid >> 5;          // wave id 0..7, owns q rows 16*wv..+15

  const long long qbase = ((long long)bh * NSEQ + (long long)w * WIN) * DIM;
  const long long kbase = ((long long)bh * NSEQ + ((long long)w - 1) * WIN) * DIM;
  const float LN1E4_32 = 9.210340371976184f / 32.0f;   // ln(10000)/(d/2)
  const float scale    = 0.125f;                       // 64^-0.5 (folded into Q)

  // ---------------- load Q (RoPE + scale) -> LDS bf16, float2 loads ---------
  for (int task = tid; task < WIN * 16; task += 256) {
    int row = task >> 4, jp = task & 15;
    int j = jp * 2;
    float t = (float)(w * WIN + row);
    float2 x1 = *(const float2*)(qg + qbase + row * DIM + j);
    float2 x2 = *(const float2*)(qg + qbase + row * DIM + j + 32);
#pragma unroll
    for (int c = 0; c < 2; ++c) {
      float ang = t * __expf(-(float)(j + c) * LN1E4_32);
      float sn, cs; __sincosf(ang, &sn, &cs);
      float a = (c ? x1.y : x1.x), b = (c ? x2.y : x2.x);
      sQ[row * STQ + j + c]      = (__bf16)((a * cs - b * sn) * scale);
      sQ[row * STQ + j + c + 32] = (__bf16)((b * cs + a * sn) * scale);
    }
  }
  // ---------------- load K (RoPE), zero the left pad ----------------
  for (int task = tid; task < KW * 16; task += 256) {
    int row = task >> 4, jp = task & 15;
    int j = jp * 2;
    int g = (w - 1) * WIN + row;              // global key position
    if (g >= 0) {
      float2 x1 = *(const float2*)(kg + kbase + row * DIM + j);
      float2 x2 = *(const float2*)(kg + kbase + row * DIM + j + 32);
#pragma unroll
      for (int c = 0; c < 2; ++c) {
        float ang = (float)g * __expf(-(float)(j + c) * LN1E4_32);
        float sn, cs; __sincosf(ang, &sn, &cs);
        float a = (c ? x1.y : x1.x), b = (c ? x2.y : x2.x);
        sK[row * STQ + j + c]      = (__bf16)(a * cs - b * sn);
        sK[row * STQ + j + c + 32] = (__bf16)(b * cs + a * sn);
      }
    } else {
#pragma unroll
      for (int c = 0; c < 2; ++c) {
        sK[row * STQ + j + c]      = (__bf16)0.f;
        sK[row * STQ + j + c + 32] = (__bf16)0.f;
      }
    }
  }
  // ---------------- load V (float4, transposed into LDS) ----------------
  for (int task = tid; task < KW * 16; task += 256) {
    int row = task >> 4, jq = task & 15;      // row = key, jq*4 = dim base
    int g = (w - 1) * WIN + row;
    float4 x = {0.f, 0.f, 0.f, 0.f};
    if (g >= 0) x = *(const float4*)(vg + kbase + row * DIM + jq * 4);
    sVt[(jq * 4 + 0) * STV + row] = (__bf16)x.x;
    sVt[(jq * 4 + 1) * STV + row] = (__bf16)x.y;
    sVt[(jq * 4 + 2) * STV + row] = (__bf16)x.z;
    sVt[(jq * 4 + 3) * STV + row] = (__bf16)x.w;
  }
  // ---------------- key validity flags (pad + attention mask) ---------------
  for (int r0 = tid; r0 < KW; r0 += 256) {
    int g = (w - 1) * WIN + r0;
    sInval[r0] = (g < 0 || amask[bidx * NSEQ + g] == 0) ? 1 : 0;
  }
  __syncthreads();

  // ---------------- QK^T : 16 N-tiles x (K=64 over 2 steps) ----------------
  const int laneHalf = lane >> 4;
  const int ln       = lane & 15;
  const int qrow     = wv * 16 + ln;          // A-matrix row (M = lane%16)

  v16bf aQ0 = gather_row(sQ, qrow, STQ, 0,  laneHalf);
  v16bf aQ1 = gather_row(sQ, qrow, STQ, 32, laneHalf);

  v8f acc[16];
#pragma unroll
  for (int nt = 0; nt < 16; ++nt) {
    v16bf b0 = gather_row(sK, nt * 16 + ln, STQ, 0,  laneHalf);
    v16bf b1 = gather_row(sK, nt * 16 + ln, STQ, 32, laneHalf);
    v8f c = {};
    c = __builtin_amdgcn_wmma_f32_16x16x32_bf16(false, aQ0, false, b0,
                                                (short)0, c, false, false);
    c = __builtin_amdgcn_wmma_f32_16x16x32_bf16(false, aQ1, false, b1,
                                                (short)0, c, false, false);
    acc[nt] = c;
  }

  // ---------------- mask, then row softmax ----------------
  // C-layout: lane ln = column within tile; VGPR r = row r (lanes<16) / 8+r.
  const float NEGMAX = -3.402823466e38f;
  float rmax[8], rsum[8];
#pragma unroll
  for (int r = 0; r < 8; ++r) rmax[r] = NEGMAX;

#pragma unroll
  for (int nt = 0; nt < 16; ++nt) {
    const int  kl      = nt * 16 + ln;               // key column 0..255
    const int  kpos    = (w - 1) * WIN + kl;         // global key position
    const bool invalid = (sInval[kl] != 0);
#pragma unroll
    for (int r = 0; r < 8; ++r) {
      int qpos = w * WIN + wv * 16 + (laneHalf ? 8 + r : r);
      bool masked = invalid || (kpos > qpos);        // pad/attn-mask or causal
      float val = masked ? NEGMAX : acc[nt][r];      // scale already in Q
      acc[nt][r] = val;
      rmax[r] = fmaxf(rmax[r], val);
    }
  }
#pragma unroll
  for (int r = 0; r < 8; ++r) {
#pragma unroll
    for (int m = 1; m < 16; m <<= 1)                 // reduce within 16-lane half
      rmax[r] = fmaxf(rmax[r], __shfl_xor(rmax[r], m, 32));
    rsum[r] = 0.f;
  }
#pragma unroll
  for (int nt = 0; nt < 16; ++nt)
#pragma unroll
    for (int r = 0; r < 8; ++r) {
      float p = __expf(acc[nt][r] - rmax[r]);        // masked -> exp(-huge)=0
      acc[nt][r] = p;
      rsum[r] += p;
    }
#pragma unroll
  for (int r = 0; r < 8; ++r) {
#pragma unroll
    for (int m = 1; m < 16; m <<= 1)
      rsum[r] += __shfl_xor(rsum[r], m, 32);
    rsum[r] = 1.0f / rsum[r];                        // keep inverse for epilogue
  }
  // write unnormalized numerators to LDS (bf16) for A-layout regather
#pragma unroll
  for (int nt = 0; nt < 16; ++nt)
#pragma unroll
    for (int r = 0; r < 8; ++r) {
      int qi = wv * 16 + (laneHalf ? 8 + r : r);
      sP[qi * STP + nt * 16 + ln] = (__bf16)acc[nt][r];
    }
  __syncthreads();

  // ---------------- P*V : 4 dim-tiles x (K=256 over 8 steps) ----------------
  // B[k][n] = V[key k][dim n]; with sVt, lane's dim row is contiguous in keys.
  v8f o[4] = {v8f{}, v8f{}, v8f{}, v8f{}};
#pragma unroll
  for (int s = 0; s < 8; ++s) {
    v16bf aP = gather_row(sP, qrow, STP, s * 32, laneHalf);
#pragma unroll
    for (int nt2 = 0; nt2 < 4; ++nt2) {
      v16bf bV = gather_row(sVt, nt2 * 16 + ln, STV, s * 32, laneHalf);
      o[nt2] = __builtin_amdgcn_wmma_f32_16x16x32_bf16(false, aP, false, bV,
                                                       (short)0, o[nt2],
                                                       false, false);
    }
  }

  // ---------------- store output (f32), normalize here ----------------
#pragma unroll
  for (int nt2 = 0; nt2 < 4; ++nt2)
#pragma unroll
    for (int r = 0; r < 8; ++r) {
      int qi = wv * 16 + (laneHalf ? 8 + r : r);
      outg[qbase + qi * DIM + nt2 * 16 + ln] = o[nt2][r] * rsum[r];
    }
}

extern "C" void kernel_launch(void* const* d_in, const int* in_sizes, int n_in,
                              void* d_out, int out_size, void* d_ws, size_t ws_size,
                              hipStream_t stream) {
  const float* q = (const float*)d_in[0];
  const float* k = (const float*)d_in[1];
  const float* v = (const float*)d_in[2];
  const int*   m = (const int*)d_in[3];
  float* out = (float*)d_out;

  dim3 grid(NSEQ / WIN, 32);   // 32 windows x (B*H = 32)
  LocalAttention_46986942218547_kernel<<<grid, 256, 0, stream>>>(q, k, v, m, out);
}
